// SpatialAttentionRefinement_4629974745214
// MI455X (gfx1250) — matrix operand order repaired and checked by reading, not compiled, not run on hardware
//
#include <hip/hip_runtime.h>

typedef __attribute__((ext_vector_type(16))) __bf16 v16bf;
typedef __attribute__((ext_vector_type(8)))  float  v8f;

#define N_SP   1152   // H*W = 48*24
#define C_DIM  1024
#define MID    256
#define BATCH  64
#define KC     64     // K chunk per LDS stage (2 WMMAs deep per C tile)
#define KPAD   66     // padded K stride (even -> dword aligned, 33 dwords -> conflict-free)
#define BM     128    // block tile rows (4 wave-rows x 32)
#define BN     128    // block tile cols (2 wave-cols x 64)

static __device__ __forceinline__ unsigned short f2bf(float f) {
  unsigned int u = __float_as_uint(f);
  u += 0x7FFFu + ((u >> 16) & 1u);          // round-to-nearest-even
  return (unsigned short)(u >> 16);
}
static __device__ __forceinline__ unsigned int pack2bf(float a, float b) {
  return (unsigned int)f2bf(a) | ((unsigned int)f2bf(b) << 16);
}

// A fragment, 16x32 bf16: lane L<16 -> row L, K+0; lanes 16-31 -> row L-16, K+8.
static __device__ __forceinline__ v16bf frag_a(const unsigned short* tile, int row, int k0, int half) {
  union { v16bf v; unsigned int u[8]; } r;
  const unsigned short* p = tile + row * KPAD + k0 + (half ? 8 : 0);
  #pragma unroll
  for (int i = 0; i < 8; ++i) {
    int kk = (i < 4) ? (2 * i) : (16 + 2 * (i - 4));
    r.u[i] = *(const unsigned int*)(p + kk);
  }
  return r.v;
}

// B fragment, 32x16 bf16 (K-major per column): lane L<16 -> col L, K+0; lanes 16-31 -> col L-16, K+8.
static __device__ __forceinline__ v16bf frag_b(const unsigned short* tile, int col, int k0, int half) {
  union { v16bf v; unsigned int u[8]; } r;
  const unsigned short* p = tile + col * KPAD + k0 + (half ? 8 : 0);
  #pragma unroll
  for (int i = 0; i < 8; ++i) {
    int kk = (i < 4) ? (2 * i) : (16 + 2 * (i - 4));
    r.u[i] = *(const unsigned int*)(p + kk);
  }
  return r.v;
}

#define WMMA_BF16(A, B, C) __builtin_amdgcn_wmma_f32_16x16x32_bf16(false, (A), false, (B), (short)0, (C), false, false)

// ---------------------------------------------------------------------------
// Kernel 1: g[b] = [w1;w2] (512x1024) @ x[b] (1024x1152), output bf16.
// grid = (N_SP/BN=9, 512/BM=4, BATCH)
// ---------------------------------------------------------------------------
__global__ __launch_bounds__(256) void k_gemm_wx(
    const float* __restrict__ w1, const float* __restrict__ w2,
    const float* __restrict__ x, unsigned short* __restrict__ g)
{
  __shared__ __align__(16) unsigned short At[BM * KPAD];
  __shared__ __align__(16) unsigned short Bt[BN * KPAD];
  const int tid = threadIdx.x;
  const int lane = tid & 31, half = lane >> 4, lm = lane & 15;
  const int wv = tid >> 5, wr = wv >> 1, wc = wv & 1;
  const int b = blockIdx.z;
  const int rowBase = blockIdx.y * BM;
  const int colBase = blockIdx.x * BN;
  v8f acc[2][4] = {};
  for (int kc = 0; kc < C_DIM; kc += KC) {
    __syncthreads();
    #pragma unroll
    for (int it = 0; it < 8; ++it) {         // A stage: 128x64, float4 along K
      int idx = tid + 256 * it;
      int k4 = (idx & 15) * 4, r = idx >> 4;
      int grow = rowBase + r;
      const float* src = (grow < MID) ? (w1 + (size_t)grow * C_DIM)
                                      : (w2 + (size_t)(grow - MID) * C_DIM);
      float4 v = *(const float4*)(src + kc + k4);
      unsigned int* dst = (unsigned int*)&At[r * KPAD + k4];
      dst[0] = pack2bf(v.x, v.y);
      dst[1] = pack2bf(v.z, v.w);
    }
    #pragma unroll
    for (int it = 0; it < 8; ++it) {         // B^T stage: Bt[n][k] = x[k][n], float4 along n
      int idx = tid + 256 * it;
      int n4 = (idx & 31) * 4, kk = idx >> 5;
      float4 v = *(const float4*)(x + ((size_t)b * C_DIM + kc + kk) * N_SP + colBase + n4);
      Bt[(n4 + 0) * KPAD + kk] = f2bf(v.x);
      Bt[(n4 + 1) * KPAD + kk] = f2bf(v.y);
      Bt[(n4 + 2) * KPAD + kk] = f2bf(v.z);
      Bt[(n4 + 3) * KPAD + kk] = f2bf(v.w);
    }
    __syncthreads();
    #pragma unroll
    for (int k0 = 0; k0 < KC; k0 += 32) {
      v16bf a0 = frag_a(At + (wr * 32 + 0)  * KPAD, lm, k0, half);
      v16bf a1 = frag_a(At + (wr * 32 + 16) * KPAD, lm, k0, half);
      #pragma unroll
      for (int j = 0; j < 4; ++j) {
        v16bf bj = frag_b(Bt + (wc * 64 + j * 16) * KPAD, lm, k0, half);
        acc[0][j] = WMMA_BF16(a0, bj, acc[0][j]);
        acc[1][j] = WMMA_BF16(a1, bj, acc[1][j]);
      }
    }
  }
  #pragma unroll
  for (int i = 0; i < 2; ++i)
    #pragma unroll
    for (int j = 0; j < 4; ++j)
      #pragma unroll
      for (int r = 0; r < 8; ++r) {          // C layout: VGPR r -> row r + 8*half, col lm
        int row = rowBase + wr * 32 + i * 16 + r + half * 8;
        int col = colBase + wc * 64 + j * 16 + lm;
        g[((size_t)b * 512 + row) * N_SP + col] = f2bf(acc[i][j][r]);
      }
}

// ---------------------------------------------------------------------------
// Kernel 2: scores[b] = g1[b]^T (1152x256) @ g2[b] (256x1152), output f32.
// grid = (N_SP/BN=9, N_SP/BM=9, BATCH)
// ---------------------------------------------------------------------------
__global__ __launch_bounds__(256) void k_scores(
    const unsigned short* __restrict__ g, float* __restrict__ attnF)
{
  __shared__ __align__(16) unsigned short At[BM * KPAD];
  __shared__ __align__(16) unsigned short Bt[BN * KPAD];
  const int tid = threadIdx.x;
  const int lane = tid & 31, half = lane >> 4, lm = lane & 15;
  const int wv = tid >> 5, wr = wv >> 1, wc = wv & 1;
  const int b = blockIdx.z;
  const int rowBase = blockIdx.y * BM;       // n
  const int colBase = blockIdx.x * BN;       // k
  const unsigned short* g1 = g + (size_t)b * 512 * N_SP;
  const unsigned short* g2 = g1 + (size_t)MID * N_SP;
  v8f acc[2][4] = {};
  for (int kc = 0; kc < MID; kc += KC) {
    __syncthreads();
    #pragma unroll
    for (int it = 0; it < 8; ++it) {         // A^T stage: At[n][m] = g1[m][n], ushort4 along n
      int idx = tid + 256 * it;
      int n4 = (idx & 31) * 4, kk = idx >> 5;
      ushort4 v = *(const ushort4*)(g1 + (size_t)(kc + kk) * N_SP + rowBase + n4);
      At[(n4 + 0) * KPAD + kk] = v.x;
      At[(n4 + 1) * KPAD + kk] = v.y;
      At[(n4 + 2) * KPAD + kk] = v.z;
      At[(n4 + 3) * KPAD + kk] = v.w;
    }
    #pragma unroll
    for (int it = 0; it < 8; ++it) {         // B^T stage: Bt[k][m] = g2[m][k], ushort4 along k
      int idx = tid + 256 * it;
      int n4 = (idx & 31) * 4, kk = idx >> 5;
      ushort4 v = *(const ushort4*)(g2 + (size_t)(kc + kk) * N_SP + colBase + n4);
      Bt[(n4 + 0) * KPAD + kk] = v.x;
      Bt[(n4 + 1) * KPAD + kk] = v.y;
      Bt[(n4 + 2) * KPAD + kk] = v.z;
      Bt[(n4 + 3) * KPAD + kk] = v.w;
    }
    __syncthreads();
    #pragma unroll
    for (int k0 = 0; k0 < KC; k0 += 32) {
      v16bf a0 = frag_a(At + (wr * 32 + 0)  * KPAD, lm, k0, half);
      v16bf a1 = frag_a(At + (wr * 32 + 16) * KPAD, lm, k0, half);
      #pragma unroll
      for (int j = 0; j < 4; ++j) {
        v16bf bj = frag_b(Bt + (wc * 64 + j * 16) * KPAD, lm, k0, half);
        acc[0][j] = WMMA_BF16(a0, bj, acc[0][j]);
        acc[1][j] = WMMA_BF16(a1, bj, acc[1][j]);
      }
    }
  }
  #pragma unroll
  for (int i = 0; i < 2; ++i)
    #pragma unroll
    for (int j = 0; j < 4; ++j)
      #pragma unroll
      for (int r = 0; r < 8; ++r) {
        int row = rowBase + wr * 32 + i * 16 + r + half * 8;
        int col = colBase + wc * 64 + j * 16 + lm;
        attnF[((size_t)b * N_SP + row) * N_SP + col] = acc[i][j][r];
      }
}

// ---------------------------------------------------------------------------
// Kernel 3: row softmax over 1152, write bf16. grid = (N_SP, BATCH)
// ---------------------------------------------------------------------------
__global__ __launch_bounds__(256) void k_softmax(
    const float* __restrict__ attnF, unsigned short* __restrict__ attnB)
{
  __shared__ float red[256];
  const int row = blockIdx.x, b = blockIdx.y, tid = threadIdx.x;
  const float* p = attnF + ((size_t)b * N_SP + row) * N_SP;
  unsigned short* q = attnB + ((size_t)b * N_SP + row) * N_SP;
  float loc[5];
  int cnt = 0;
  float m = -3.4e38f;
  for (int i = tid; i < N_SP; i += 256) { float v = p[i]; loc[cnt++] = v; m = fmaxf(m, v); }
  red[tid] = m; __syncthreads();
  for (int s = 128; s > 0; s >>= 1) { if (tid < s) red[tid] = fmaxf(red[tid], red[tid + s]); __syncthreads(); }
  m = red[0]; __syncthreads();
  float sum = 0.f;
  for (int j = 0; j < cnt; ++j) { loc[j] = __expf(loc[j] - m); sum += loc[j]; }
  red[tid] = sum; __syncthreads();
  for (int s = 128; s > 0; s >>= 1) { if (tid < s) red[tid] += red[tid + s]; __syncthreads(); }
  const float inv = 1.0f / red[0];
  cnt = 0;
  for (int i = tid; i < N_SP; i += 256) { q[i] = f2bf(loc[cnt++] * inv); }
}

// ---------------------------------------------------------------------------
// Kernel 4: out[b] = 0.001*alpha * (cam[b] (1024x1152) @ attn[b] (1152x1152)) + cam[b]
// grid = (N_SP/BN=9, C_DIM/BM=8, BATCH)
// ---------------------------------------------------------------------------
__global__ __launch_bounds__(256) void k_refine(
    const float* __restrict__ cam, const unsigned short* __restrict__ attnB,
    const float* __restrict__ alpha_p, float* __restrict__ out)
{
  __shared__ __align__(16) unsigned short At[BM * KPAD];
  __shared__ __align__(16) unsigned short Bt[BN * KPAD];
  const int tid = threadIdx.x;
  const int lane = tid & 31, half = lane >> 4, lm = lane & 15;
  const int wv = tid >> 5, wr = wv >> 1, wc = wv & 1;
  const int b = blockIdx.z;
  const int rowBase = blockIdx.y * BM;       // c
  const int colBase = blockIdx.x * BN;       // i
  v8f acc[2][4] = {};
  for (int kc = 0; kc < N_SP; kc += KC) {
    if (kc + KC < N_SP)                      // prefetch next attn chunk (global_prefetch_b8)
      __builtin_prefetch(attnB + ((size_t)b * N_SP + kc + KC + (tid >> 5)) * N_SP + colBase + (tid & 31) * 4, 0, 0);
    __syncthreads();
    #pragma unroll
    for (int it = 0; it < 8; ++it) {         // A stage: cam rows, float4 along j
      int idx = tid + 256 * it;
      int k4 = (idx & 15) * 4, r = idx >> 4;
      float4 v = *(const float4*)(cam + ((size_t)b * C_DIM + rowBase + r) * N_SP + kc + k4);
      unsigned int* dst = (unsigned int*)&At[r * KPAD + k4];
      dst[0] = pack2bf(v.x, v.y);
      dst[1] = pack2bf(v.z, v.w);
    }
    #pragma unroll
    for (int it = 0; it < 8; ++it) {         // B^T stage: Bt[i][j] = attn[j][i], ushort4 along i
      int idx = tid + 256 * it;
      int n4 = (idx & 31) * 4, kk = idx >> 5;
      ushort4 v = *(const ushort4*)(attnB + ((size_t)b * N_SP + kc + kk) * N_SP + colBase + n4);
      Bt[(n4 + 0) * KPAD + kk] = v.x;
      Bt[(n4 + 1) * KPAD + kk] = v.y;
      Bt[(n4 + 2) * KPAD + kk] = v.z;
      Bt[(n4 + 3) * KPAD + kk] = v.w;
    }
    __syncthreads();
    #pragma unroll
    for (int k0 = 0; k0 < KC; k0 += 32) {
      v16bf a0 = frag_a(At + (wr * 32 + 0)  * KPAD, lm, k0, half);
      v16bf a1 = frag_a(At + (wr * 32 + 16) * KPAD, lm, k0, half);
      #pragma unroll
      for (int j = 0; j < 4; ++j) {
        v16bf bj = frag_b(Bt + (wc * 64 + j * 16) * KPAD, lm, k0, half);
        acc[0][j] = WMMA_BF16(a0, bj, acc[0][j]);
        acc[1][j] = WMMA_BF16(a1, bj, acc[1][j]);
      }
    }
  }
  const float scale = 0.001f * alpha_p[0];
  #pragma unroll
  for (int i = 0; i < 2; ++i)
    #pragma unroll
    for (int j = 0; j < 4; ++j)
      #pragma unroll
      for (int r = 0; r < 8; ++r) {
        int row = rowBase + wr * 32 + i * 16 + r + half * 8;
        int col = colBase + wc * 64 + j * 16 + lm;
        size_t o = ((size_t)b * C_DIM + row) * N_SP + col;
        out[o] = scale * acc[i][j][r] + cam[o];
      }
}

// ---------------------------------------------------------------------------
extern "C" void kernel_launch(void* const* d_in, const int* in_sizes, int n_in,
                              void* d_out, int out_size, void* d_ws, size_t ws_size,
                              hipStream_t stream) {
  (void)in_sizes; (void)n_in; (void)out_size; (void)ws_size;
  const float* feat  = (const float*)d_in[0];
  const float* cam   = (const float*)d_in[1];
  const float* w1    = (const float*)d_in[2];
  const float* w2    = (const float*)d_in[3];
  const float* alpha = (const float*)d_in[4];
  float* out = (float*)d_out;

  char* ws = (char*)d_ws;
  const size_t gBytes     = (size_t)BATCH * 512 * N_SP * 2;        //  72 MB bf16 g1|g2
  const size_t attnFBytes = (size_t)BATCH * N_SP * N_SP * 4;       // 324 MB f32 scores
  unsigned short* g     = (unsigned short*)ws;
  float*          attnF = (float*)(ws + gBytes);
  unsigned short* attnB = (unsigned short*)(ws + gBytes + attnFBytes);

  dim3 blk(256);
  k_gemm_wx<<<dim3(N_SP / BN, 512 / BM, BATCH), blk, 0, stream>>>(w1, w2, feat, g);
  k_scores <<<dim3(N_SP / BN, N_SP / BM, BATCH), blk, 0, stream>>>(g, attnF);
  k_softmax<<<dim3(N_SP, BATCH), blk, 0, stream>>>(attnF, attnB);
  k_refine <<<dim3(N_SP / BN, C_DIM / BM, BATCH), blk, 0, stream>>>(cam, attnB, alpha, out);
}
